// SelfAttention_14551349198882
// MI455X (gfx1250) — 16-bit and 8-bit matrix operand order repaired and checked by reading, compiled, not run on hardware
//
#include <hip/hip_runtime.h>
#include <hip/hip_bf16.h>

#define BB   8
#define CC   256
#define NN   4096
#define CQK  32
#define SSTR 68

typedef __attribute__((ext_vector_type(16))) __bf16 v16bf;
typedef __attribute__((ext_vector_type(8)))  float  v8f;
typedef __attribute__((ext_vector_type(8)))  int    v8i;

union BV  { uint4 u[2]; v16bf v; };
union B8  { unsigned int w[8]; uint4 u[2]; v8i v; };

__device__ __forceinline__ unsigned short f2bf(float f) {
    unsigned int u = __float_as_uint(f);
    unsigned int r = (u + 0x7FFFu + ((u >> 16) & 1u)) >> 16;
    return (unsigned short)r;
}

__device__ __forceinline__ unsigned char f2fp8_sw(float f) {
    float cf = fminf(fmaxf(f, -448.f), 448.f);
    unsigned int u = __float_as_uint(cf);
    unsigned int s = (u >> 24) & 0x80u;
    int e = (int)((u >> 23) & 0xFFu) - 127 + 7;
    unsigned int m = (u >> 20) & 0x7u;
    if (e <= 0) return (unsigned char)s;
    if (e > 15) { e = 15; m = 6; }
    return (unsigned char)(s | ((unsigned)e << 3) | m);
}

__device__ __forceinline__ unsigned char f2fp8(float f) {
#if __has_builtin(__builtin_amdgcn_cvt_pk_fp8_f32)
    return (unsigned char)(__builtin_amdgcn_cvt_pk_fp8_f32(f, f, 0, false) & 0xFF);
#else
    return f2fp8_sw(f);
#endif
}

__device__ __forceinline__ unsigned int pack4_fp8(float a, float b, float c, float d) {
#if __has_builtin(__builtin_amdgcn_cvt_pk_fp8_f32)
    int w = __builtin_amdgcn_cvt_pk_fp8_f32(a, b, 0, false);
    w     = __builtin_amdgcn_cvt_pk_fp8_f32(c, d, w, true);
    return (unsigned int)w;
#else
    return (unsigned)f2fp8_sw(a) | ((unsigned)f2fp8_sw(b) << 8) |
           ((unsigned)f2fp8_sw(c) << 16) | ((unsigned)f2fp8_sw(d) << 24);
#endif
}

__global__ __launch_bounds__(256) void qk_proj_kernel(
    const float* __restrict__ x,
    const float* __restrict__ Wq, const float* __restrict__ bq,
    const float* __restrict__ Wk, const float* __restrict__ bk,
    unsigned short* __restrict__ qb, unsigned short* __restrict__ kb)
{
    const int t    = threadIdx.x;
    const int p    = t & 63;
    const int g    = t >> 6;
    const int b    = blockIdx.x >> 6;
    const int tile = blockIdx.x & 63;
    const int n    = tile * 64 + p;

    const float* W    = (g < 2 ? Wq : Wk) + (size_t)(g & 1) * 16 * CC;
    const float* bias = (g < 2 ? bq : bk) + (g & 1) * 16;

    float acc[16];
#pragma unroll
    for (int o = 0; o < 16; ++o) acc[o] = 0.f;

    const float* xp = x + (size_t)b * CC * NN + n;
#pragma unroll 4
    for (int c = 0; c < CC; ++c) {
        float xv = xp[(size_t)c * NN];
#pragma unroll
        for (int o = 0; o < 16; ++o) acc[o] += W[o * CC + c] * xv;
    }

    unsigned int pk[8];
#pragma unroll
    for (int i = 0; i < 8; ++i) {
        unsigned int lo = f2bf(acc[2 * i]     + bias[2 * i]);
        unsigned int hi = f2bf(acc[2 * i + 1] + bias[2 * i + 1]);
        pk[i] = lo | (hi << 16);
    }
    unsigned short* dst = (g < 2 ? qb : kb) + ((size_t)b * NN + n) * CQK + (g & 1) * 16;
    *(uint4*)(dst)     = make_uint4(pk[0], pk[1], pk[2], pk[3]);
    *(uint4*)(dst + 8) = make_uint4(pk[4], pk[5], pk[6], pk[7]);
}

__global__ __launch_bounds__(256) void v_proj_kernel(
    const float* __restrict__ x,
    const float* __restrict__ Wv, const float* __restrict__ bv,
    unsigned char* __restrict__ vt)
{
    const int t    = threadIdx.x;
    const int p    = t & 31;
    const int g    = t >> 5;
    const int b    = blockIdx.x >> 7;
    const int tile = blockIdx.x & 127;
    const int n    = tile * 32 + p;

    const float* W = Wv + (size_t)g * 32 * CC;
    float acc[32];
#pragma unroll
    for (int o = 0; o < 32; ++o) acc[o] = 0.f;

    const float* xp = x + (size_t)b * CC * NN + n;
#pragma unroll 2
    for (int c = 0; c < CC; ++c) {
        float xv = xp[(size_t)c * NN];
#pragma unroll
        for (int o = 0; o < 32; ++o) acc[o] += W[o * CC + c] * xv;
    }
#pragma unroll
    for (int o = 0; o < 32; ++o) {
        float v = acc[o] + bv[g * 32 + o];
        vt[((size_t)b * CC + g * 32 + o) * NN + n] = f2fp8(v);
    }
}

__global__ __launch_bounds__(256) void attn_kernel(
    const unsigned short* __restrict__ qb,
    const unsigned short* __restrict__ kb,
    const unsigned char*  __restrict__ vt,
    const float* __restrict__ x,
    const float* __restrict__ gamma,
    float* __restrict__ out)
{
    __shared__ float ssh[8][16 * SSTR];
    __shared__ float astat[8][16];
    __shared__ float lstat[8][16];

    const int tid  = threadIdx.x;
    const int lane = tid & 31;
    const int wave = tid >> 5;
    const int b    = blockIdx.x >> 6;
    const int qg   = blockIdx.x & 63;
    const int qt   = qg * 4 + (wave >> 1);
    const int ch   = wave & 1;
    const int q0   = qt * 16;

    const int hi    = lane >> 4;
    const int lx    = lane & 15;
    const int koffe = hi ? 8 : 0;

    BV qa;
    const unsigned short* qp = qb + ((size_t)b * NN + q0 + lx) * CQK;
    qa.u[0] = *(const uint4*)(qp + koffe);
    qa.u[1] = *(const uint4*)(qp + 16 + koffe);

    float mrow = -1e30f, lrow = 0.f;
    v8f o_[8];
#pragma unroll
    for (int i = 0; i < 8; ++i) o_[i] = (v8f){};

    const unsigned short* kbase = kb + (size_t)b * NN * CQK;
    const unsigned char*  vbase = vt + (size_t)b * CC * NN;
    float*       swr  = &ssh[wave][0];
    const float* srow = &ssh[wave][lx * SSTR];

    for (int jc = 0; jc < NN; jc += 64) {
        BV kbt[4];
#pragma unroll
        for (int u = 0; u < 4; ++u) {
            const unsigned short* kp =
                kbase + (size_t)(jc + u * 16 + lx) * CQK + (hi ? 8 : 0);
            kbt[u].u[0] = *(const uint4*)(kp);
            kbt[u].u[1] = *(const uint4*)(kp + 16);
        }
        if (jc + 64 < NN) {
            __builtin_prefetch(kbase + (size_t)(jc + 64 + lx) * CQK, 0, 1);
            __builtin_prefetch(kbase + (size_t)(jc + 96 + lx) * CQK, 0, 1);
        }

        v8f zz = {};
        v8f st[4];
#pragma unroll
        for (int u = 0; u < 4; ++u)
            st[u] = __builtin_amdgcn_wmma_f32_16x16x32_bf16(
                        false, qa.v, false, kbt[u].v, (short)0, zz, false, false);

#pragma unroll
        for (int i = 0; i < 8; ++i) {
            const int r = i + (hi << 3);
#pragma unroll
            for (int u = 0; u < 4; ++u)
                swr[r * SSTR + u * 16 + lx] = st[u][i];
        }

        float s[32];
#pragma unroll
        for (int u = 0; u < 4; ++u) {
            float4 a = *(const float4*)(srow + (u >> 1) * 32 + hi * 16 + (u & 1) * 8);
            float4 c = *(const float4*)(srow + (u >> 1) * 32 + hi * 16 + (u & 1) * 8 + 4);
            s[8 * u + 0] = a.x; s[8 * u + 1] = a.y; s[8 * u + 2] = a.z; s[8 * u + 3] = a.w;
            s[8 * u + 4] = c.x; s[8 * u + 5] = c.y; s[8 * u + 6] = c.z; s[8 * u + 7] = c.w;
        }

        float mx = s[0];
#pragma unroll
        for (int j = 1; j < 32; ++j) mx = fmaxf(mx, s[j]);
        mx = fmaxf(mx, __shfl_xor(mx, 16));

        const float mn = fmaxf(mrow, mx);
        const float al = __expf(mrow - mn);

        float rs = 0.f;
        B8 pa;
#pragma unroll
        for (int v = 0; v < 8; ++v) {
            float p0 = __expf(s[4 * v]     - mn);
            float p1 = __expf(s[4 * v + 1] - mn);
            float p2 = __expf(s[4 * v + 2] - mn);
            float p3 = __expf(s[4 * v + 3] - mn);
            rs += (p0 + p1) + (p2 + p3);
            pa.w[v] = pack4_fp8(p0, p1, p2, p3);
        }
        rs += __shfl_xor(rs, 16);
        lrow = lrow * al + rs;
        mrow = mn;

        if (hi == 0) astat[wave][lx] = al;
        float4 a0 = *(const float4*)(&astat[wave][hi << 3]);
        float4 a1 = *(const float4*)(&astat[wave][(hi << 3) + 4]);
        const float af[8] = { a0.x, a0.y, a0.z, a0.w, a1.x, a1.y, a1.z, a1.w };
#pragma unroll
        for (int t = 0; t < 8; ++t)
#pragma unroll
            for (int i = 0; i < 8; ++i) o_[t][i] *= af[i];

#pragma unroll
        for (int t = 0; t < 8; ++t) {
            const int c = ch * 128 + t * 16 + lx;
            const unsigned char* vp = vbase + (size_t)c * NN + jc + (hi ? 16 : 0);
            B8 vb;
            vb.u[0] = *(const uint4*)(vp);
            vb.u[1] = *(const uint4*)(vp + 32);
            o_[t] = __builtin_amdgcn_wmma_f32_16x16x64_fp8_fp8(
                        pa.v, vb.v, (short)0, o_[t], false, false);
        }
    }

    if (hi == 0) lstat[wave][lx] = lrow;
    float4 l0 = *(const float4*)(&lstat[wave][hi << 3]);
    float4 l1 = *(const float4*)(&lstat[wave][(hi << 3) + 4]);
    const float lv[8] = { l0.x, l0.y, l0.z, l0.w, l1.x, l1.y, l1.z, l1.w };
    float inv[8];
#pragma unroll
    for (int i = 0; i < 8; ++i) inv[i] = 1.f / lv[i];

    const float gm = gamma[0];
#pragma unroll
    for (int t = 0; t < 8; ++t) {
        const int c = ch * 128 + t * 16 + lx;
        const size_t base = ((size_t)b * CC + c) * NN + q0 + (hi << 3);
        const float* xp = x + base;
        float* op = out + base;
#pragma unroll
        for (int i = 0; i < 8; ++i)
            op[i] = gm * (o_[t][i] * inv[i]) + xp[i];
    }
}

extern "C" void kernel_launch(void* const* d_in, const int* in_sizes, int n_in,
                              void* d_out, int out_size, void* d_ws, size_t ws_size,
                              hipStream_t stream) {
    const float* x  = (const float*)d_in[0];
    const float* Wq = (const float*)d_in[1];
    const float* bq = (const float*)d_in[2];
    const float* Wk = (const float*)d_in[3];
    const float* bk = (const float*)d_in[4];
    const float* Wv = (const float*)d_in[5];
    const float* bv = (const float*)d_in[6];
    const float* gm = (const float*)d_in[7];
    float* out = (float*)d_out;

    unsigned short* qbuf = (unsigned short*)d_ws;
    unsigned short* kbuf = qbuf + (size_t)BB * NN * CQK;
    unsigned char*  vbuf = (unsigned char*)(kbuf + (size_t)BB * NN * CQK);

    qk_proj_kernel<<<BB * (NN / 64), 256, 0, stream>>>(x, Wq, bq, Wk, bk, qbuf, kbuf);
    v_proj_kernel<<<BB * (NN / 32), 256, 0, stream>>>(x, Wv, bv, vbuf);
    attn_kernel<<<BB * 64, 256, 0, stream>>>(qbuf, kbuf, vbuf, x, gm, out);
}
